// TinySpeechZ_61220463837781
// MI455X (gfx1250) — matrix operand order repaired and checked by reading, not compiled, not run on hardware
//
#include <hip/hip_runtime.h>
#include <hip/hip_bf16.h>
#include <hip/hip_fp16.h>

typedef __attribute__((ext_vector_type(16))) _Float16 v16h_t;
typedef __attribute__((ext_vector_type(8)))  _Float16 v8h_t;
typedef __attribute__((ext_vector_type(8)))  float    v8f_t;

union V16H { v16h_t v; v8h_t q[2]; _Float16 h[16]; };
union V8F  { v8f_t  v; float    f[8];  };

#define NIMG    128
#define CMAIN   7
#define FR      128
#define HR      64
#define PLANE_F (FR*FR)   // 16384
#define PLANE_H (HR*HR)   // 4096

// ---------------------------------------------------------------------------
// One-time weight packer: builds the per-lane V_WMMA_F32_16X16X32_F16 B-operand
// register images (zero-padded), plus zero-padded bias vectors.
// B layout (32x16 KxN): col = lane&15, elems 0..7 -> K=khalf*8+e,
// elems 8..15 -> K=16+khalf*8+(e-8). B[k][n] = W[n*cin + k].
// pk: 4 matrices x 512 halfs = [B1a, B1b, B2, B3]
// pkb: [gb padded to 32][pb padded to 16][eb padded to 16] floats
// ---------------------------------------------------------------------------
__global__ void pack_weights_kernel(
    const float* __restrict__ gw, const float* __restrict__ gb,
    const float* __restrict__ pw, const float* __restrict__ pb,
    const float* __restrict__ ew, const float* __restrict__ eb,
    int mid, int out, _Float16* __restrict__ pk, float* __restrict__ pkb)
{
    const int t = threadIdx.x;           // 64 threads
    if (t < 32) {
        const int col = t & 15, khalf = t >> 4;
        for (int e = 0; e < 16; ++e) {
            const int K = (e < 8) ? (khalf * 8 + e) : (16 + khalf * 8 + (e - 8));
            pk[0 * 512 + t * 16 + e] =
                (col < mid && K < CMAIN) ? (_Float16)gw[col * CMAIN + K] : (_Float16)0.0f;
            pk[1 * 512 + t * 16 + e] =
                ((col + 16) < mid && K < CMAIN) ? (_Float16)gw[(col + 16) * CMAIN + K] : (_Float16)0.0f;
            pk[2 * 512 + t * 16 + e] =
                (col < out && K < mid) ? (_Float16)pw[col * mid + K] : (_Float16)0.0f;
            pk[3 * 512 + t * 16 + e] =
                (col < CMAIN && K < out) ? (_Float16)ew[col * out + K] : (_Float16)0.0f;
        }
        pkb[t] = (t < mid) ? gb[t] : 0.0f;
    } else {
        const int u = t - 32;            // 0..31
        if (u < 16) {
            pkb[32 + u] = (u < out)   ? pb[u] : 0.0f;
            pkb[48 + u] = (u < CMAIN) ? eb[u] : 0.0f;
        }
    }
}

// ---------------------------------------------------------------------------
// Fused attention condenser (branch-free hot path):
//   pending BN affine + 2x2 maxpool + (7->mid, relu) + (mid->out, relu)
//   + (out->7) + sigmoid  ==> S[img][7][64][64]
// One wave = tile of 16 half-res pixels; 4 x v_wmma_f32_16x16x32_f16.
// Per-wave LDS tile (zero-padded 16x32 f16) stages pixel-major activations;
// A operands are two ds_load_b128 each; B operands two global_load_b128 each.
// ---------------------------------------------------------------------------
__global__ __launch_bounds__(256) void condenser_attn_kernel(
    const float* __restrict__ X, float* __restrict__ S,
    const _Float16* __restrict__ pk, const float* __restrict__ pkb,
    const float* __restrict__ affine)
{
    __shared__ __align__(32) _Float16 stage[8][16][32];
    const int wave  = threadIdx.x >> 5;
    const int lane  = threadIdx.x & 31;
    const int tile  = blockIdx.x * 8 + wave;     // 32768 tiles total
    const int img   = tile >> 8;
    const int tin   = tile & 255;
    const int col   = lane & 15;
    const int khalf = lane >> 4;
    const int mbase = khalf * 8;                 // C/D row base
    _Float16* sw = &stage[wave][0][0];

    // B operands + padded biases (all unconditional, cache-hot)
    const v16h_t b1a = *reinterpret_cast<const v16h_t*>(pk + 0 * 512 + lane * 16);
    const v16h_t b1b = *reinterpret_cast<const v16h_t*>(pk + 1 * 512 + lane * 16);
    const v16h_t b2  = *reinterpret_cast<const v16h_t*>(pk + 2 * 512 + lane * 16);
    const v16h_t b3  = *reinterpret_cast<const v16h_t*>(pk + 3 * 512 + lane * 16);
    const float bias1a = pkb[col];
    const float bias1b = pkb[col + 16];
    const float bias2  = pkb[32 + col];
    const float bias3  = pkb[48 + col];

    // ---- zero the stage tile (covers channels >= 7 for the A1 build) ----
    {
        V16H z;
#pragma unroll
        for (int i = 0; i < 16; ++i) z.h[i] = (_Float16)0.0f;
        *reinterpret_cast<v16h_t*>(sw + lane * 16) = z.v;
    }
    __syncthreads();

    // ---- maxpool2(affine(x)): all 32 lanes load; pixel m = col,
    //      lane-group 0 -> channels 0..3, group 1 -> channels 4..6 ----
    {
        const int p  = tin * 16 + col;
        const int ph = p >> 6, pwd = p & 63;
#pragma unroll
        for (int i = 0; i < 4; ++i) {
            const int c = khalf * 4 + i;
            if (c < CMAIN) {
                const float al = affine[c], be = affine[8 + c];
                const float* xp = X + ((size_t)(img * CMAIN + c) * FR + 2 * ph) * FR + 2 * pwd;
                const float2 v01 = *reinterpret_cast<const float2*>(xp);
                const float2 v23 = *reinterpret_cast<const float2*>(xp + FR);
                float q0 = al * v01.x + be, q1 = al * v01.y + be;
                float q2 = al * v23.x + be, q3 = al * v23.y + be;
                stage[wave][col][c] = (_Float16)fmaxf(fmaxf(q0, q1), fmaxf(q2, q3));
            }
        }
    }
    __syncthreads();

    // ---- A1 (16x32 f16 A layout: elems 0..7 -> K=khalf*8+e, 8..15 -> 16+khalf*8+e) ----
    V16H a1;
    a1.q[0] = *reinterpret_cast<const v8h_t*>(sw + col * 32 + khalf * 8);
    a1.q[1] = *reinterpret_cast<const v8h_t*>(sw + col * 32 + 16 + khalf * 8);

    const v8f_t zeroC = {0.f, 0.f, 0.f, 0.f, 0.f, 0.f, 0.f, 0.f};

    // ---- layer 1: 7 -> mid (two N-tiles, second is zero matrix if mid<=16) ----
    V8F c1a; c1a.v = __builtin_amdgcn_wmma_f32_16x16x32_f16(false, a1.v, false, b1a,
                                                            (short)0, zeroC, false, false);
    V8F c1b; c1b.v = __builtin_amdgcn_wmma_f32_16x16x32_f16(false, a1.v, false, b1b,
                                                            (short)0, zeroC, false, false);
    __syncthreads();
#pragma unroll
    for (int r = 0; r < 8; ++r) {
        stage[wave][mbase + r][col]      = (_Float16)fmaxf(c1a.f[r] + bias1a, 0.0f);
        stage[wave][mbase + r][col + 16] = (_Float16)fmaxf(c1b.f[r] + bias1b, 0.0f);
    }
    __syncthreads();

    // ---- A2, layer 2: mid -> out (out <= 14) ----
    V16H a2;
    a2.q[0] = *reinterpret_cast<const v8h_t*>(sw + col * 32 + khalf * 8);
    a2.q[1] = *reinterpret_cast<const v8h_t*>(sw + col * 32 + 16 + khalf * 8);
    V8F c2; c2.v = __builtin_amdgcn_wmma_f32_16x16x32_f16(false, a2.v, false, b2,
                                                          (short)0, zeroC, false, false);
    __syncthreads();
#pragma unroll
    for (int r = 0; r < 8; ++r) {
        stage[wave][mbase + r][col]      = (_Float16)fmaxf(c2.f[r] + bias2, 0.0f);
        stage[wave][mbase + r][col + 16] = (_Float16)0.0f;
    }
    __syncthreads();

    // ---- A3, layer 3: out -> 7, sigmoid -> S ----
    V16H a3;
    a3.q[0] = *reinterpret_cast<const v8h_t*>(sw + col * 32 + khalf * 8);
    a3.q[1] = *reinterpret_cast<const v8h_t*>(sw + col * 32 + 16 + khalf * 8);
    V8F c3; c3.v = __builtin_amdgcn_wmma_f32_16x16x32_f16(false, a3.v, false, b3,
                                                          (short)0, zeroC, false, false);
    if (col < CMAIN) {
        float* sp = S + (size_t)(img * CMAIN + col) * PLANE_H + tin * 16;
#pragma unroll
        for (int r = 0; r < 8; ++r) {
            float aval = c3.f[r] + bias3;
            sp[mbase + r] = 1.0f / (1.0f + __expf(-aval));
        }
    }
}

// ---------------------------------------------------------------------------
// x = (alpha*x+beta) * (1 + S_up2 * scale); accumulate per-channel sum/sumsq.
// One block = 1024 consecutive elements of one (img, channel) plane.
// ---------------------------------------------------------------------------
__global__ __launch_bounds__(256) void apply_stats_kernel(
    float* __restrict__ X, const float* __restrict__ S,
    const float* __restrict__ affine, const float* __restrict__ scale,
    float* __restrict__ stats)
{
    __shared__ float rsum[256], rsq[256];
    const int blk   = blockIdx.x;          // 14336
    const int plane = blk >> 4;            // img*7 + c
    const int chunk = blk & 15;
    const int c     = plane % CMAIN;
    const float al = affine[c], be = affine[8 + c], sc = scale[0];
    const int base = plane * PLANE_F + chunk * 1024 + threadIdx.x * 4;

    float4 xv = *reinterpret_cast<const float4*>(X + base);
    const int p = base & (PLANE_F - 1);
    const int h = p >> 7, w = p & 127;     // w multiple of 4
    const float* sp = S + (size_t)plane * PLANE_H + (h >> 1) * HR + (w >> 1);
    const float s0 = sp[0], s1 = sp[1];

    float y0 = (al * xv.x + be) * (1.0f + s0 * sc);
    float y1 = (al * xv.y + be) * (1.0f + s0 * sc);
    float y2 = (al * xv.z + be) * (1.0f + s1 * sc);
    float y3 = (al * xv.w + be) * (1.0f + s1 * sc);
    float4 yv = {y0, y1, y2, y3};
    *reinterpret_cast<float4*>(X + base) = yv;

    rsum[threadIdx.x] = y0 + y1 + y2 + y3;
    rsq[threadIdx.x]  = y0 * y0 + y1 * y1 + y2 * y2 + y3 * y3;
    __syncthreads();
    for (int off = 128; off > 0; off >>= 1) {
        if (threadIdx.x < off) {
            rsum[threadIdx.x] += rsum[threadIdx.x + off];
            rsq[threadIdx.x]  += rsq[threadIdx.x + off];
        }
        __syncthreads();
    }
    if (threadIdx.x == 0) {
        atomicAdd(&stats[c],     rsum[0]);
        atomicAdd(&stats[8 + c], rsq[0]);
    }
}

// stats -> pending affine: alpha = g*rsqrt(var+eps), beta = b - mean*alpha
__global__ void bn_finalize_kernel(const float* __restrict__ stats,
                                   const float* __restrict__ g, const float* __restrict__ b,
                                   float* __restrict__ affine_out)
{
    int c = threadIdx.x;
    if (c < CMAIN) {
        const float inv_cnt = 1.0f / (128.0f * 128.0f * 128.0f);
        float m  = stats[c] * inv_cnt;
        float v  = stats[8 + c] * inv_cnt - m * m;
        float al = g[c] * rsqrtf(v + 1e-5f);
        affine_out[c]     = al;
        affine_out[8 + c] = b[c] - m * al;
    }
}

__global__ void init_ws_kernel(float* __restrict__ stats, float* __restrict__ pool,
                               float* __restrict__ affine0)
{
    int i = blockIdx.x * 256 + threadIdx.x;
    if (i < 12 * 16)    stats[i] = 0.0f;
    if (i < NIMG * 17)  pool[i]  = 0.0f;
    if (i < 16)         affine0[i] = (i < 8) ? 1.0f : 0.0f;
}

// conv1: 1->7, 3x3, pad 1, relu. One thread = one spatial pixel (all 7 oc).
__global__ __launch_bounds__(256) void conv1_kernel(
    const float* __restrict__ X0, const float* __restrict__ W,
    const float* __restrict__ B, float* __restrict__ X)
{
    __shared__ float ws[63], bs[7];
    if (threadIdx.x < 63) ws[threadIdx.x] = W[threadIdx.x];
    if (threadIdx.x < 7)  bs[threadIdx.x] = B[threadIdx.x];
    __syncthreads();
    const int idx = blockIdx.x * 256 + threadIdx.x;
    const int img = idx >> 14;
    const int p   = idx & (PLANE_F - 1);
    const int h = p >> 7, w = p & 127;
    float patch[9];
#pragma unroll
    for (int dy = -1; dy <= 1; ++dy)
#pragma unroll
        for (int dx = -1; dx <= 1; ++dx) {
            int hh = h + dy, ww = w + dx;
            bool ok = (hh >= 0) && (hh < FR) && (ww >= 0) && (ww < FR);
            patch[(dy + 1) * 3 + (dx + 1)] =
                ok ? X0[(size_t)img * PLANE_F + hh * FR + ww] : 0.0f;
        }
#pragma unroll
    for (int oc = 0; oc < CMAIN; ++oc) {
        float acc = bs[oc];
#pragma unroll
        for (int k = 0; k < 9; ++k) acc += ws[oc * 9 + k] * patch[k];
        X[(size_t)(img * CMAIN + oc) * PLANE_F + p] = fmaxf(acc, 0.0f);
    }
}

// conv2 (7->17, 3x3, pad 1, relu) fused with global average pool accumulation.
__global__ __launch_bounds__(256) void conv2_pool_kernel(
    const float* __restrict__ X, const float* __restrict__ W2,
    const float* __restrict__ B2, const float* __restrict__ affine,
    float* __restrict__ pool)
{
    __shared__ float ws[17 * 63];
    __shared__ float red[256];
    for (int i = threadIdx.x; i < 17 * 63; i += 256) ws[i] = W2[i];
    __syncthreads();
    const int img = blockIdx.x >> 6;                 // 64 blocks per image
    const int p   = (blockIdx.x & 63) * 256 + threadIdx.x;
    const int h = p >> 7, w = p & 127;
    float patch[63];
#pragma unroll
    for (int c = 0; c < CMAIN; ++c) {
        const float al = affine[c], be = affine[8 + c];
#pragma unroll
        for (int dy = -1; dy <= 1; ++dy)
#pragma unroll
            for (int dx = -1; dx <= 1; ++dx) {
                int hh = h + dy, ww = w + dx;
                bool ok = (hh >= 0) && (hh < FR) && (ww >= 0) && (ww < FR);
                float v = ok ? X[(size_t)(img * CMAIN + c) * PLANE_F + hh * FR + ww] : 0.0f;
                patch[c * 9 + (dy + 1) * 3 + (dx + 1)] = ok ? (al * v + be) : 0.0f;
            }
    }
    for (int oc = 0; oc < 17; ++oc) {
        float acc = B2[oc];
#pragma unroll
        for (int k = 0; k < 63; ++k) acc += ws[oc * 63 + k] * patch[k];
        acc = fmaxf(acc, 0.0f);
        red[threadIdx.x] = acc;
        __syncthreads();
        for (int off = 128; off > 0; off >>= 1) {
            if (threadIdx.x < off) red[threadIdx.x] += red[threadIdx.x + off];
            __syncthreads();
        }
        if (threadIdx.x == 0) atomicAdd(&pool[img * 17 + oc], red[0]);
        __syncthreads();
    }
}

__global__ void fc_kernel(const float* __restrict__ pool,
                          const float* __restrict__ fw, const float* __restrict__ fb,
                          float* __restrict__ out)
{
    int i = blockIdx.x * 256 + threadIdx.x;
    if (i < NIMG * 35) {
        int n = i / 35, k = i % 35;
        float acc = fb[k];
#pragma unroll
        for (int c = 0; c < 17; ++c)
            acc += (pool[n * 17 + c] * (1.0f / 16384.0f)) * fw[k * 17 + c];
        out[i] = acc;
    }
}

// ---------------------------------------------------------------------------
// Host orchestration
// ---------------------------------------------------------------------------
struct CondP { const float *gw, *gb, *pw, *pb, *ew, *eb, *scale; int mid, out; };
struct BNP   { const float *g, *b; };

static const int MID0[6] = {14, 14, 14, 14, 14, 14};
static const int OUT0[6] = {3, 3, 2, 11, 14, 10};
static const int MID1[6] = {6, 6, 4, 22, 28, 20};
static const int OUT1[6] = {7, 7, 7, 7, 7, 7};

extern "C" void kernel_launch(void* const* d_in, const int* in_sizes, int n_in,
                              void* d_out, int out_size, void* d_ws, size_t ws_size,
                              hipStream_t stream)
{
    (void)n_in; (void)out_size; (void)ws_size;
    auto F = [&](int i) { return (const float*)d_in[i]; };
    const float *x0, *c1w, *c1b, *c2w, *c2b, *fw, *fb;
    CondP cond[12];
    BNP   bn[12];

    // Detect flattening order: dict insertion order puts x (2,097,152 elems)
    // first; jax sorted-key tree flattening puts blocks[0].bn1_b (7) first.
    const bool insertion = (in_sizes[0] == 2097152);
    if (insertion) {
        int idx = 0;
        x0 = F(idx++); c1w = F(idx++); c1b = F(idx++);
        for (int bI = 0; bI < 6; ++bI) {
            for (int j = 0; j < 2; ++j) {
                CondP& c = cond[bI * 2 + j];
                c.gw = F(idx++); c.gb = F(idx++); c.pw = F(idx++); c.pb = F(idx++);
                c.ew = F(idx++); c.eb = F(idx++); c.scale = F(idx++);
            }
            bn[bI * 2 + 0].g = F(idx++); bn[bI * 2 + 0].b = F(idx++);
            bn[bI * 2 + 1].g = F(idx++); bn[bI * 2 + 1].b = F(idx++);
        }
        c2w = F(idx++); c2b = F(idx++); fw = F(idx++); fb = F(idx++);
    } else {
        // jax tree_leaves: dict keys sorted alphabetically, 'params' < 'x'
        int idx = 0;
        for (int bI = 0; bI < 6; ++bI) {
            bn[bI * 2 + 0].b = F(idx++); bn[bI * 2 + 0].g = F(idx++);
            bn[bI * 2 + 1].b = F(idx++); bn[bI * 2 + 1].g = F(idx++);
            for (int j = 0; j < 2; ++j) {
                CondP& c = cond[bI * 2 + j];
                c.eb = F(idx++); c.ew = F(idx++); c.gb = F(idx++); c.gw = F(idx++);
                c.pb = F(idx++); c.pw = F(idx++); c.scale = F(idx++);
            }
        }
        c1b = F(idx++); c1w = F(idx++); c2b = F(idx++); c2w = F(idx++);
        fb  = F(idx++); fw  = F(idx++); x0  = F(idx++);
    }
    for (int i = 0; i < 6; ++i) {
        cond[i * 2 + 0].mid = MID0[i]; cond[i * 2 + 0].out = OUT0[i];
        cond[i * 2 + 1].mid = MID1[i]; cond[i * 2 + 1].out = OUT1[i];
    }

    // Workspace layout (all regions 256B-aligned)
    char* wp = (char*)d_ws;
    float*     X      = (float*)wp;     wp += sizeof(float) * (size_t)NIMG * CMAIN * PLANE_F; // 58.7 MB
    float*     S      = (float*)wp;     wp += sizeof(float) * (size_t)NIMG * CMAIN * PLANE_H; // 14.7 MB
    _Float16*  packW  = (_Float16*)wp;  wp += sizeof(_Float16) * 12 * 2048;                   // 48 KB
    float*     packB  = (float*)wp;     wp += sizeof(float) * 12 * 64;                        // 3 KB
    float*     stats  = (float*)wp;     wp += sizeof(float) * 12 * 16;
    float*     affine = (float*)wp;     wp += 1024;
    float*     pool   = (float*)wp;     wp += sizeof(float) * NIMG * 17;

    init_ws_kernel<<<9, 256, 0, stream>>>(stats, pool, affine);
    for (int ci = 0; ci < 12; ++ci) {
        const CondP& c = cond[ci];
        pack_weights_kernel<<<1, 64, 0, stream>>>(
            c.gw, c.gb, c.pw, c.pb, c.ew, c.eb, c.mid, c.out,
            packW + ci * 2048, packB + ci * 64);
    }
    conv1_kernel<<<(NIMG * PLANE_F) / 256, 256, 0, stream>>>(x0, c1w, c1b, X);

    for (int ci = 0; ci < 12; ++ci) {
        const CondP& c = cond[ci];
        condenser_attn_kernel<<<4096, 256, 0, stream>>>(
            X, S, packW + ci * 2048, packB + ci * 64, affine + ci * 16);
        apply_stats_kernel<<<14336, 256, 0, stream>>>(
            X, S, affine + ci * 16, c.scale, stats + ci * 16);
        bn_finalize_kernel<<<1, 32, 0, stream>>>(
            stats + ci * 16, bn[ci].g, bn[ci].b, affine + (ci + 1) * 16);
    }

    conv2_pool_kernel<<<NIMG * 64, 256, 0, stream>>>(X, c2w, c2b, affine + 12 * 16, pool);
    fc_kernel<<<(NIMG * 35 + 255) / 256, 256, 0, stream>>>(pool, fw, fb, (float*)d_out);
}
